// GRACENew2_73821897883939
// MI455X (gfx1250) — matrix operand order repaired and checked by reading, not compiled, not run on hardware
//
#include <hip/hip_runtime.h>
#include <hip/hip_bf16.h>

// GRACE contrastive loss on MI455X (gfx1250, wave32, WMMA + TDM).
// N=16384, HIDDEN=512, PROJ=128, TAU=0.5 -> 1/TAU = 2.0
// Pipeline:
//   k_project_norm : u=ELU(z W1+b1) W2+b2, fused L2-normalize, emit f16 nu/nv
//   k_gram_loss    : TDM double-buffered j-tiles in LDS; f16 WMMA gram tiles;
//                    exp + half-wave reductions -> per-row loss terms
//   k_reduce       : mean over rows -> scalar

#define NROWS 16384
#define HID   512
#define PROJ  128
#define KC    64

typedef _Float16 h16v __attribute__((ext_vector_type(16)));
typedef _Float16 h8v  __attribute__((ext_vector_type(8)));
typedef float    f8v  __attribute__((ext_vector_type(8)));
typedef unsigned u32x4 __attribute__((ext_vector_type(4)));
typedef unsigned u32x8 __attribute__((ext_vector_type(8)));

union HU { h16v v; h8v h[2]; };

__device__ __forceinline__ f8v wmma_f16(h16v a, h16v b, f8v c) {
  // D = A(16x32 f16) * B(32x16 f16) + C(16x16 f32)
  return __builtin_amdgcn_wmma_f32_16x16x32_f16(
      /*neg_a=*/false, a, /*neg_b=*/false, b,
      /*c_mod=*/(short)0, c, /*reuse_a=*/false, /*reuse_b=*/false);
}

__device__ __forceinline__ f8v zero8() {
  f8v r;
#pragma unroll
  for (int i = 0; i < 8; ++i) r[i] = 0.f;
  return r;
}

// ---------------------------------------------------------------------------
// Tensor Data Mover: 1-D contiguous copy global -> LDS (nbytes, 4B elements).
// Builds a 2-group D# (count=1, type=2 "image"); issued per-wave, EXEC
// ignored; tracked by TENSORcnt. Inline asm form is portable across the
// ROCm-7.2 (5-arg builtin) and therock (6-arg builtin) toolchains.
// ---------------------------------------------------------------------------
__device__ __forceinline__ void tdm_load_1d(const void* lds_dst, const void* gsrc,
                                            unsigned nbytes) {
  const unsigned dw = nbytes >> 2;   // 4-byte elements
  const unsigned long long ga = (unsigned long long)gsrc;
  const unsigned lo = (unsigned)(unsigned long long)lds_dst;  // LDS byte offset

  u32x4 g0;
  g0[0] = 1u;                                   // count=1 (valid user descriptor)
  g0[1] = lo;                                   // lds_addr [63:32]
  g0[2] = (unsigned)ga;                         // global_addr [95:64]
  g0[3] = ((unsigned)(ga >> 32) & 0x01FFFFFFu)  // global_addr [120:96]
          | (2u << 30);                         // type=2 [127:126]

  u32x8 g1;
  g1[0] = 2u << 16;                 // workgroup_mask=0, data_size=2 (4 bytes)
  g1[1] = (dw & 0xFFFFu) << 16;     // tensor_dim0[15:0] @ bits 63:48
  g1[2] = ((dw >> 16) & 0xFFFFu)    // tensor_dim0[31:16]
          | (1u << 16);             // tensor_dim1 = 1  (bits 111:80)
  g1[3] = (dw & 0xFFFFu) << 16;     // tile_dim0 @ bits 127:112
  g1[4] = 0u;                       // tile_dim1=0 (unused), tile_dim2=0
  g1[5] = dw;                       // tensor_dim0_stride[31:0]
  g1[6] = 0u;                       // stride hi / dim1 stride lo
  g1[7] = 0u;

  asm volatile("tensor_load_to_lds %0, %1" :: "s"(g0), "s"(g1) : "memory");
}

// ---------------------------------------------------------------------------
// Kernel 1: projection head + fused L2 normalize, f16 output.
// Grid: 256 blocks (blk<128 -> z1/u rows, else z2/v rows), 256 threads (8 waves).
// Each wave computes 16 output rows x 128 cols.
// Dynamic LDS (64 KB): phase A: zA[128x64] + w1T[128x64] f16;
//                      phase B: hA[128x128] (reuses A) + w2T[128x128] f16.
// ---------------------------------------------------------------------------
extern "C" __global__ __launch_bounds__(256)
void k_project_norm(const float* __restrict__ z1, const float* __restrict__ z2,
                    const float* __restrict__ W1, const float* __restrict__ b1,
                    const float* __restrict__ W2, const float* __restrict__ b2,
                    _Float16* __restrict__ nuv /* [2N x 128] */) {
  extern __shared__ _Float16 smem[];
  _Float16* zA  = smem;               // 128*64
  _Float16* w1T = smem + 128 * KC;    // 128*64  (transposed: [p][k])
  _Float16* hA  = smem;               // 128*128 (reuses phase-A region)
  _Float16* w2T = smem + 128 * 128;   // 128*128 (transposed: [pout][pin])

  const int tid = threadIdx.x;
  const int wv  = tid >> 5;
  const int l   = tid & 31;
  const int lm  = l & 15;   // lane within half-wave
  const int lh  = l >> 4;   // half-wave select
  const int blk = blockIdx.x;

  const float* zsrc = (blk < 128) ? (z1 + (size_t)blk * 128 * HID)
                                  : (z2 + (size_t)(blk - 128) * 128 * HID);
  const size_t orow0 = (size_t)blk * 128;  // output row base in the 2N space

  f8v acc[8];
#pragma unroll
  for (int t = 0; t < 8; ++t) acc[t] = zero8();

  // ---------------- GEMM1: h = z @ W1, K = 512 in chunks of 64 -------------
  for (int kc = 0; kc < HID / KC; ++kc) {
    for (int i = tid; i < 128 * KC; i += 256) {
      int r = i >> 6, c = i & (KC - 1);
      zA[r * KC + c] = (_Float16)zsrc[(size_t)r * HID + kc * KC + c];
    }
    for (int i = tid; i < KC * PROJ; i += 256) {
      int k = i >> 7, p = i & 127;  // coalesced read over p
      w1T[p * KC + k] = (_Float16)W1[(size_t)(kc * KC + k) * PROJ + p];
    }
    if (kc + 1 < HID / KC) {
      // pull next K-chunk toward the caches while we compute (HBM streams)
      __builtin_prefetch(&zsrc[(size_t)(tid >> 1) * HID + (kc + 1) * KC + (tid & 1) * 32], 0, 1);
      __builtin_prefetch(&W1[(size_t)((kc + 1) * KC + (tid >> 2)) * PROJ + (tid & 3) * 32], 0, 1);
    }
    __syncthreads();

    const int arow = wv * 16 + lm;
#pragma unroll
    for (int kch = 0; kch < KC; kch += 32) {
      HU aU;
      const _Float16* ab = &zA[arow * KC + kch + 8 * lh];
      aU.h[0] = *(const h8v*)ab;          // K = base..base+7
      aU.h[1] = *(const h8v*)(ab + 16);   // K = base+16..base+23
      HU bU[8];
#pragma unroll
      for (int nt = 0; nt < 8; ++nt) {    // batch all B frags, then WMMA chain
        const _Float16* bb = &w1T[(nt * 16 + lm) * KC + kch + 8 * lh];
        bU[nt].h[0] = *(const h8v*)bb;
        bU[nt].h[1] = *(const h8v*)(bb + 16);
      }
#pragma unroll
      for (int nt = 0; nt < 8; ++nt)
        acc[nt] = wmma_f16(aU.v, bU[nt].v, acc[nt]);
    }
    __syncthreads();
  }

  // ---------------- bias + ELU -> hA (f16), stage W2^T ---------------------
#pragma unroll
  for (int nt = 0; nt < 8; ++nt) {
    const int col = nt * 16 + lm;
    const float bb1 = b1[col];
#pragma unroll
    for (int g = 0; g < 8; ++g) {
      float x = acc[nt][g] + bb1;
      x = (x > 0.f) ? x : (__expf(x) - 1.f);          // ELU, alpha=1
      hA[(wv * 16 + g + 8 * lh) * PROJ + col] = (_Float16)x;
    }
  }
  for (int i = tid; i < PROJ * PROJ; i += 256) {
    int k = i >> 7, p = i & 127;
    w2T[p * PROJ + k] = (_Float16)W2[(size_t)k * PROJ + p];
  }
  __syncthreads();

  // ---------------- GEMM2: u = h @ W2 + b2, K = 128 ------------------------
  f8v acc2[8];
#pragma unroll
  for (int nt = 0; nt < 8; ++nt) {
    const float bb2 = b2[nt * 16 + lm];
#pragma unroll
    for (int g = 0; g < 8; ++g) acc2[nt][g] = bb2;
  }
  const int arow2 = wv * 16 + lm;
#pragma unroll
  for (int c2 = 0; c2 < PROJ; c2 += 32) {
    HU aU;
    const _Float16* ab = &hA[arow2 * PROJ + c2 + 8 * lh];
    aU.h[0] = *(const h8v*)ab;
    aU.h[1] = *(const h8v*)(ab + 16);
    HU bU[8];
#pragma unroll
    for (int nt = 0; nt < 8; ++nt) {
      const _Float16* bb = &w2T[(nt * 16 + lm) * PROJ + c2 + 8 * lh];
      bU[nt].h[0] = *(const h8v*)bb;
      bU[nt].h[1] = *(const h8v*)(bb + 16);
    }
#pragma unroll
    for (int nt = 0; nt < 8; ++nt)
      acc2[nt] = wmma_f16(aU.v, bU[nt].v, acc2[nt]);
  }

  // ---------------- fused L2 normalize + f16 store -------------------------
#pragma unroll
  for (int g = 0; g < 8; ++g) {
    float ss = 0.f;
#pragma unroll
    for (int nt = 0; nt < 8; ++nt) ss += acc2[nt][g] * acc2[nt][g];
    ss += __shfl_xor(ss, 1, 32);
    ss += __shfl_xor(ss, 2, 32);
    ss += __shfl_xor(ss, 4, 32);
    ss += __shfl_xor(ss, 8, 32);
    const float inv = 1.f / fmaxf(sqrtf(ss), 1e-12f);
    const size_t row = orow0 + wv * 16 + g + 8 * lh;
#pragma unroll
    for (int nt = 0; nt < 8; ++nt)
      nuv[row * PROJ + nt * 16 + lm] = (_Float16)(acc2[nt][g] * inv);
  }
}

// ---------------------------------------------------------------------------
// Kernel 2: gram tiles + softmax denominators.
// Grid: 128 blocks x 256 threads. Wave owns 16 rows i, block sweeps 1024
// j-tiles. Each j-tile (nu 4KB + nv 4KB) is staged ONCE per block into LDS by
// the Tensor Data Mover (wave 0 issues 2 descriptors), double-buffered so the
// DMA of tile j+1 overlaps the 16 WMMAs + exp/reduce of tile j.
// ---------------------------------------------------------------------------
extern "C" __global__ __launch_bounds__(256)
void k_gram_loss(const _Float16* __restrict__ nuv, float* __restrict__ row_terms) {
  __shared__ _Float16 jbuf[2][2][16 * PROJ];   // [buffer][u/v][16x128] = 16 KB

  const int tid = threadIdx.x;
  const int wv  = tid >> 5;
  const int l   = tid & 31;
  const int lm  = l & 15;
  const int lh  = l >> 4;
  const size_t rbase = (size_t)blockIdx.x * 128 + wv * 16;

  const _Float16* nu = nuv;
  const _Float16* nv = nuv + (size_t)NROWS * PROJ;
  constexpr int NT = NROWS / 16;
  constexpr unsigned TILE_BYTES = 16 * PROJ * sizeof(_Float16);  // 4096

  // Loop-invariant A fragments: rows rbase..rbase+15 of nu and nv.
  HU au[4], av[4];
  {
    const _Float16* pa = &nu[(rbase + lm) * PROJ];
    const _Float16* pv = &nv[(rbase + lm) * PROJ];
#pragma unroll
    for (int c = 0; c < 4; ++c) {
      const int o = c * 32 + 8 * lh;
      au[c].h[0] = *(const h8v*)(pa + o);
      au[c].h[1] = *(const h8v*)(pa + o + 16);
      av[c].h[0] = *(const h8v*)(pv + o);
      av[c].h[1] = *(const h8v*)(pv + o + 16);
    }
  }

  float D1[8], D2[8], duu[8], dvv[8], pos[8];
#pragma unroll
  for (int g = 0; g < 8; ++g) { D1[g] = 0.f; D2[g] = 0.f; duu[g] = 0.f; dvv[g] = 0.f; pos[g] = 1.f; }

  const int diag_jt = (int)(rbase >> 4);

  // Preload tile 0 into buffer 0.
  if (wv == 0) {
    tdm_load_1d(&jbuf[0][0][0], nu, TILE_BYTES);
    tdm_load_1d(&jbuf[0][1][0], nv, TILE_BYTES);
  }

  for (int jt = 0; jt < NT; ++jt) {
    const int cur = jt & 1;
    if (wv == 0) __builtin_amdgcn_s_wait_tensorcnt(0);  // tile jt landed in LDS
    __syncthreads();                                    // publish to all waves
    if (wv == 0 && jt + 1 < NT) {                       // DMA next tile (other buffer)
      tdm_load_1d(&jbuf[cur ^ 1][0][0], nu + (size_t)(jt + 1) * 16 * PROJ, TILE_BYTES);
      tdm_load_1d(&jbuf[cur ^ 1][1][0], nv + (size_t)(jt + 1) * 16 * PROJ, TILE_BYTES);
    }

    // B fragments from LDS: columns j of the gram = rows j of nu/nv; K contiguous.
    const _Float16* pu = &jbuf[cur][0][lm * PROJ];
    const _Float16* pv = &jbuf[cur][1][lm * PROJ];
    HU bu[4], bv[4];
#pragma unroll
    for (int c = 0; c < 4; ++c) {
      const int o = c * 32 + 8 * lh;
      bu[c].h[0] = *(const h8v*)(pu + o);
      bu[c].h[1] = *(const h8v*)(pu + o + 16);
      bv[c].h[0] = *(const h8v*)(pv + o);
      bv[c].h[1] = *(const h8v*)(pv + o + 16);
    }

    f8v cuu = zero8(), cuv = zero8(), cvv = zero8(), cvu = zero8();
#pragma unroll
    for (int c = 0; c < 4; ++c) {
      cuu = wmma_f16(au[c].v, bu[c].v, cuu);
      cuv = wmma_f16(au[c].v, bv[c].v, cuv);
      cvv = wmma_f16(av[c].v, bv[c].v, cvv);
      cvu = wmma_f16(av[c].v, bu[c].v, cvu);
    }

    const bool isdiag = (jt == diag_jt);
#pragma unroll
    for (int g = 0; g < 8; ++g) {
      // 1/TAU = 2
      float euu = __expf(2.f * cuu[g]);
      float euv = __expf(2.f * cuv[g]);
      float evv = __expf(2.f * cvv[g]);
      float evu = __expf(2.f * cvu[g]);
      if (isdiag) {
        // Diagonal (m,m) lives at lane (m<8 ? m : m+16), VGPR g=m%8.
        // Representative lanes (lm==0) fetch rows g+8*lh -> src = 24*lh+g.
        const int src = 24 * lh + g;
        duu[g] = __shfl(euu, src, 32);
        dvv[g] = __shfl(evv, src, 32);
        pos[g] = __shfl(euv, src, 32);
      }
      // Row sums across the 16 lanes of a half-wave (N dimension).
      float s1 = euu + euv;   // denom1 = S1 + S2
      float s2 = evv + evu;   // denom2 = S3 + S4
      s1 += __shfl_xor(s1, 1, 32); s1 += __shfl_xor(s1, 2, 32);
      s1 += __shfl_xor(s1, 4, 32); s1 += __shfl_xor(s1, 8, 32);
      s2 += __shfl_xor(s2, 1, 32); s2 += __shfl_xor(s2, 2, 32);
      s2 += __shfl_xor(s2, 4, 32); s2 += __shfl_xor(s2, 8, 32);
      D1[g] += s1;
      D2[g] += s2;
    }
  }

  if (lm == 0) {
#pragma unroll
    for (int g = 0; g < 8; ++g) {
      const size_t row = rbase + g + 8 * lh;
      const float d1 = D1[g] - duu[g];   // intra_u + inter - diag(intra_u)
      const float d2 = D2[g] - dvv[g];   // intra_v + inter^T - diag(intra_v)
      row_terms[row] = 0.5f * (__logf(d1) + __logf(d2)) - __logf(pos[g]);
    }
  }
}

// ---------------------------------------------------------------------------
// Kernel 3: mean over N rows -> scalar.
// ---------------------------------------------------------------------------
extern "C" __global__ __launch_bounds__(256)
void k_reduce(const float* __restrict__ row_terms, float* __restrict__ out) {
  __shared__ float red[256];
  float s = 0.f;
  for (int i = threadIdx.x; i < NROWS; i += 256) s += row_terms[i];
  red[threadIdx.x] = s;
  __syncthreads();
  for (int st = 128; st > 0; st >>= 1) {
    if (threadIdx.x < st) red[threadIdx.x] += red[threadIdx.x + st];
    __syncthreads();
  }
  if (threadIdx.x == 0) out[0] = red[0] / (float)NROWS;
}

// ---------------------------------------------------------------------------
extern "C" void kernel_launch(void* const* d_in, const int* in_sizes, int n_in,
                              void* d_out, int out_size, void* d_ws, size_t ws_size,
                              hipStream_t stream) {
  const float* z1 = (const float*)d_in[0];
  const float* z2 = (const float*)d_in[1];
  const float* W1 = (const float*)d_in[2];
  const float* b1 = (const float*)d_in[3];
  const float* W2 = (const float*)d_in[4];
  const float* b2 = (const float*)d_in[5];

  char* ws = (char*)d_ws;
  _Float16* nuv = (_Float16*)ws;                              // 2N*128 f16 = 8 MiB
  float* row_terms = (float*)(ws + (size_t)2 * NROWS * PROJ * sizeof(_Float16));

  k_project_norm<<<256, 256, 65536, stream>>>(z1, z2, W1, b1, W2, b2, nuv);
  k_gram_loss<<<128, 256, 0, stream>>>(nuv, row_terms);
  k_reduce<<<1, 256, 0, stream>>>(row_terms, (float*)d_out);
}
